// NTM_317827580326
// MI455X (gfx1250) — matrix operand order repaired and checked by reading, not compiled, not run on hardware
//
#include <hip/hip_runtime.h>

typedef __attribute__((ext_vector_type(16))) __bf16 v16bf;
typedef __attribute__((ext_vector_type(8)))  float  v8f;

#define BATCH 65536
#define DDIM 64
#define KDIM 100
#define KPAD 112          // 7 tiles of 16
#define NT 7
#define CDIM 5
#define ROWS_PER_BLOCK 128
#define THREADS 256

union AFrag { unsigned u[8]; v16bf v; };

__device__ __forceinline__ unsigned f2bf(float f) {
    unsigned u = __builtin_bit_cast(unsigned, f);
    return (u + 0x7FFFu + ((u >> 16) & 1u)) >> 16;   // RNE truncation to bf16
}
__device__ __forceinline__ unsigned pack_bf2(float a, float b) {
    return f2bf(a) | (f2bf(b) << 16);
}

__global__ __launch_bounds__(THREADS, 2)
void ntm_fused(const int* __restrict__ i_in, const int* __restrict__ j_in,
               const int* __restrict__ k_in,
               const float* __restrict__ user_emb, const float* __restrict__ item_emb,
               const float* __restrict__ time_emb,
               const float* __restrict__ w,
               const float* __restrict__ U1, const float* __restrict__ U2,
               const float* __restrict__ U3,
               const float* __restrict__ fc_w, const float* __restrict__ fc_b,
               float* __restrict__ out)
{
    __shared__ unsigned wt32[KPAD * 32];            // W^T bf16 [N=112][K=64], packed K-pairs
    __shared__ unsigned h32[ROWS_PER_BLOCK * 32];   // Hadamard bf16 [128][64], packed pairs
    __shared__ float u1s[DDIM * CDIM], u2s[DDIM * CDIM], u3s[DDIM * CDIM];
    __shared__ float fcwg[KPAD];                    // fc_w[0:100] padded with zeros
    __shared__ float fcwc[CDIM * CDIM * CDIM];      // fc_w[100:225]
    __shared__ float abc[ROWS_PER_BLOCK * 15];      // a,b,c per row
    __shared__ float rowsum[ROWS_PER_BLOCK];        // tucker + bias per row

    const int tid  = threadIdx.x;
    const int wave = tid >> 5;
    const int lane = tid & 31;
    const int hf   = lane >> 4;     // which 16-lane half
    const int l16  = lane & 15;

    // ---- Phase 1: stage constants into LDS ----
    for (int idx = tid; idx < KPAD * 32; idx += THREADS) {
        int nn = idx >> 5, kp = idx & 31;
        int k0 = kp * 2;
        float v0 = (nn < KDIM) ? w[k0 * KDIM + nn]       : 0.f;
        float v1 = (nn < KDIM) ? w[(k0 + 1) * KDIM + nn] : 0.f;
        wt32[idx] = pack_bf2(v0, v1);
    }
    for (int idx = tid; idx < DDIM * CDIM; idx += THREADS) {
        u1s[idx] = U1[idx]; u2s[idx] = U2[idx]; u3s[idx] = U3[idx];
    }
    if (tid < KPAD) fcwg[tid] = (tid < KDIM) ? fc_w[tid] : 0.f;
    if (tid < 125)  fcwc[tid] = fc_w[KDIM + tid];
    __syncthreads();

    // ---- Phase 2: coalesced gathers, Hadamard -> bf16 LDS, mode-product partials ----
    const int rowbase = blockIdx.x * ROWS_PER_BLOCK;
    for (int iter = 0; iter < 8; ++iter) {
        int r  = wave * 16 + iter * 2 + hf;     // 2 rows per wave per iteration
        int gr = rowbase + r;
        size_t iu = (size_t)i_in[gr] * DDIM;
        size_t ju = (size_t)j_in[gr] * DDIM;
        size_t ku = (size_t)k_in[gr] * DDIM;
        float4 ie = *(const float4*)(user_emb + iu + l16 * 4);
        float4 je = *(const float4*)(item_emb + ju + l16 * 4);
        float4 ke = *(const float4*)(time_emb + ku + l16 * 4);
        float ia[4] = {ie.x, ie.y, ie.z, ie.w};
        float ja[4] = {je.x, je.y, je.z, je.w};
        float ka[4] = {ke.x, ke.y, ke.z, ke.w};
        float h[4];
        #pragma unroll
        for (int e = 0; e < 4; ++e) h[e] = ia[e] * ja[e] * ka[e];
        h32[r * 32 + l16 * 2 + 0] = pack_bf2(h[0], h[1]);
        h32[r * 32 + l16 * 2 + 1] = pack_bf2(h[2], h[3]);

        float pa[5], pb[5], pc[5];
        #pragma unroll
        for (int c = 0; c < 5; ++c) { pa[c] = 0.f; pb[c] = 0.f; pc[c] = 0.f; }
        #pragma unroll
        for (int e = 0; e < 4; ++e) {
            int eb = (l16 * 4 + e) * CDIM;
            #pragma unroll
            for (int c = 0; c < 5; ++c) {
                pa[c] += ia[e] * u1s[eb + c];
                pb[c] += ja[e] * u2s[eb + c];
                pc[c] += ka[e] * u3s[eb + c];
            }
        }
        #pragma unroll
        for (int m = 8; m >= 1; m >>= 1) {   // reduce across the 16 lanes of this row
            #pragma unroll
            for (int c = 0; c < 5; ++c) {
                pa[c] += __shfl_xor(pa[c], m, 32);
                pb[c] += __shfl_xor(pb[c], m, 32);
                pc[c] += __shfl_xor(pc[c], m, 32);
            }
        }
        if (l16 == 0) {
            #pragma unroll
            for (int c = 0; c < 5; ++c) {
                abc[r * 15 + c]      = pa[c];
                abc[r * 15 + 5 + c]  = pb[c];
                abc[r * 15 + 10 + c] = pc[c];
            }
        }
    }
    __syncthreads();

    // ---- Phase 3: per-row Tucker 5x5x5 dot + bias ----
    if (tid < ROWS_PER_BLOCK) {
        float av[5], bv[5], cv[5];
        #pragma unroll
        for (int c = 0; c < 5; ++c) {
            av[c] = abc[tid * 15 + c];
            bv[c] = abc[tid * 15 + 5 + c];
            cv[c] = abc[tid * 15 + 10 + c];
        }
        float t = 0.f;
        #pragma unroll
        for (int x = 0; x < 5; ++x) {
            #pragma unroll
            for (int y = 0; y < 5; ++y) {
                float axy = av[x] * bv[y];
                #pragma unroll
                for (int z = 0; z < 5; ++z)
                    t += axy * cv[z] * fcwc[(x * 5 + y) * 5 + z];
            }
        }
        rowsum[tid] = t + fc_b[0];
    }
    __syncthreads();

    // ---- Phase 4: WMMA [16x64]x[64x112] + ReLU + fc_w dot, per wave ----
    // A fragment (16x32 bf16): lane l: M=l&15; VGPR v holds K = 2*(v&3)+8*half+16*(v>>2), K+1
    AFrag alo, ahi;
    {
        int base = (wave * 16 + l16) * 32;
        #pragma unroll
        for (int v = 0; v < 8; ++v) {
            int ks = (v & 3) + 4 * hf + 8 * (v >> 2);   // K>>1
            alo.u[v] = h32[base + ks];
            ahi.u[v] = h32[base + 16 + ks];
        }
    }
    float acc[8];
    #pragma unroll
    for (int v = 0; v < 8; ++v) acc[v] = 0.f;

    #pragma unroll
    for (int t = 0; t < NT; ++t) {
        // B fragment (32x16 bf16): lane l: N=l&15; VGPR v holds K = 2*(v&3)+8*half+16*(v>>2), K+1
        AFrag blo, bhi;
        #pragma unroll
        for (int v = 0; v < 8; ++v) {
            int nidx = (t * 16 + l16) * 32 + (v & 3) + 4 * hf + 8 * (v >> 2);
            blo.u[v] = wt32[nidx];
            bhi.u[v] = wt32[nidx + 16];
        }
        v8f c8 = {0.f, 0.f, 0.f, 0.f, 0.f, 0.f, 0.f, 0.f};
        c8 = __builtin_amdgcn_wmma_f32_16x16x32_bf16(false, alo.v, false, blo.v,
                                                     (short)0, c8, false, false);
        c8 = __builtin_amdgcn_wmma_f32_16x16x32_bf16(false, ahi.v, false, bhi.v,
                                                     (short)0, c8, false, false);
        float fw = fcwg[t * 16 + l16];
        #pragma unroll
        for (int v = 0; v < 8; ++v)
            acc[v] += fmaxf(c8[v], 0.f) * fw;   // relu * fc_w[N]
    }

    // reduce over N (16 lanes per half); D layout: lane l holds M = v + 8*half
    #pragma unroll
    for (int m = 8; m >= 1; m >>= 1) {
        #pragma unroll
        for (int v = 0; v < 8; ++v)
            acc[v] += __shfl_xor(acc[v], m, 32);
    }

    if (l16 == 0) {
        #pragma unroll
        for (int v = 0; v < 8; ++v) {
            int rloc = wave * 16 + 8 * hf + v;
            out[rowbase + rloc] = acc[v] + rowsum[rloc];
        }
    }
}

extern "C" void kernel_launch(void* const* d_in, const int* in_sizes, int n_in,
                              void* d_out, int out_size, void* d_ws, size_t ws_size,
                              hipStream_t stream) {
    const int*   i_in     = (const int*)d_in[0];
    const int*   j_in     = (const int*)d_in[1];
    const int*   k_in     = (const int*)d_in[2];
    const float* user_emb = (const float*)d_in[3];
    const float* item_emb = (const float*)d_in[4];
    const float* time_emb = (const float*)d_in[5];
    const float* w        = (const float*)d_in[6];
    const float* U1       = (const float*)d_in[7];
    const float* U2       = (const float*)d_in[8];
    const float* U3       = (const float*)d_in[9];
    const float* fc_w     = (const float*)d_in[10];
    const float* fc_b     = (const float*)d_in[11];
    float* out = (float*)d_out;

    dim3 grid(BATCH / ROWS_PER_BLOCK);   // 512 blocks
    dim3 block(THREADS);                 // 8 waves (wave32)
    ntm_fused<<<grid, block, 0, stream>>>(i_in, j_in, k_in, user_emb, item_emb,
                                          time_emb, w, U1, U2, U3, fc_w, fc_b, out);
}
